// RadarP4Transformer_35794257445567
// MI455X (gfx1250) — matrix operand order repaired and checked by reading, not compiled, not run on hardware
//
#include <hip/hip_runtime.h>
#include <hip/hip_bf16.h>
#include <math.h>

// ---------------- problem constants (from reference) ----------------
constexpr int kB      = 4;
constexpr int kT      = 24;
constexpr int kN      = 4096;
constexpr int kTO     = 12;      // temporal frames after stride
constexpr int kM      = 128;     // anchors per frame (N / 32)
constexpr int kDIM    = 512;
constexpr int kSEQ    = kTO * kM;        // 1536 tokens
constexpr int kHEADS  = 8;
constexpr int kDH     = 64;
constexpr int kMLP    = 1024;
constexpr int kNC     = 30;
constexpr int kNS     = 32;              // NSAMPLES
constexpr float kRAD2 = 0.3f * 0.3f;
constexpr int kDEPTH  = 4;

typedef __bf16 bf16;
typedef __attribute__((ext_vector_type(16))) __bf16 v16bf;
typedef __attribute__((ext_vector_type(8)))  __bf16 v8bf;
typedef __attribute__((ext_vector_type(8)))  float  v8f;
typedef __attribute__((ext_vector_type(4)))  unsigned int u32x4;
typedef __attribute__((ext_vector_type(8)))  int i32x8;
typedef __attribute__((ext_vector_type(4)))  int i32x4;

#if __has_builtin(__builtin_amdgcn_tensor_load_to_lds) && __has_builtin(__builtin_amdgcn_s_wait_tensorcnt)
#define HAVE_TDM 1
#else
#define HAVE_TDM 0
#endif

__device__ __forceinline__ float gelu_exact(float x) {
    return 0.5f * x * (1.0f + erff(x * 0.70710678118654752f));
}

// 16-lane XOR butterfly step: v_permlane16_b32 (VALU) if available, else shfl
__device__ __forceinline__ float xor16_step(float v, unsigned lo, unsigned hi, int m) {
#if __has_builtin(__builtin_amdgcn_permlane16)
    int iv = __builtin_bit_cast(int, v);
    iv = __builtin_amdgcn_permlane16(iv, iv, (int)lo, (int)hi, false, false);
    return __builtin_bit_cast(float, iv);
#else
    return __shfl_xor(v, m, 16);
#endif
}
__device__ __forceinline__ float redmax16(float v) {
    v = fmaxf(v, xor16_step(v, 0x67452301u, 0xEFCDAB89u, 1));
    v = fmaxf(v, xor16_step(v, 0x54761032u, 0xDCFE98BAu, 2));
    v = fmaxf(v, xor16_step(v, 0x32107654u, 0xBA98FEDCu, 4));
    v = fmaxf(v, xor16_step(v, 0xFEDCBA98u, 0x76543210u, 8));
    return v;
}
__device__ __forceinline__ float redsum16(float v) {
    v += xor16_step(v, 0x67452301u, 0xEFCDAB89u, 1);
    v += xor16_step(v, 0x54761032u, 0xDCFE98BAu, 2);
    v += xor16_step(v, 0x32107654u, 0xBA98FEDCu, 4);
    v += xor16_step(v, 0xFEDCBA98u, 0x76543210u, 8);
    return v;
}

#if HAVE_TDM
// TDM: async-copy a 2D tile (rows x rowlen bf16, row stride in elements) to LDS.
__device__ __forceinline__ void tdm_load_2d(const bf16* gptr, unsigned lds_off,
                                            unsigned rows, unsigned rowlen,
                                            unsigned row_stride_elems) {
    unsigned long long ga = (unsigned long long)(size_t)gptr;
    u32x4 g0;
    g0[0] = 1u;                                             // count=1 (valid user D#)
    g0[1] = lds_off;                                        // lds_addr (bytes)
    g0[2] = (unsigned)(ga & 0xFFFFFFFFu);                   // global_addr[31:0]
    g0[3] = (unsigned)((ga >> 32) & 0x1FFFFFFu) | (2u << 30); // addr[56:32] | type=2
    i32x8 g1;
    g1[0] = (int)(1u << 16);                                // data_size = 2 bytes
    g1[1] = (int)(rowlen << 16);                            // tensor_dim0[15:0]
    g1[2] = (int)((rowlen >> 16) | (rows << 16));           // dim0 hi | tensor_dim1 lo
    g1[3] = (int)((rows >> 16) | (rowlen << 16));           // dim1 hi | tile_dim0
    g1[4] = (int)(rows & 0xFFFFu);                          // tile_dim1 (tile_dim2 = 0)
    g1[5] = (int)row_stride_elems;                          // tensor_dim0_stride[31:0]
    g1[6] = 0;
    g1[7] = 0;
    i32x4 z4 = {0, 0, 0, 0};
#if defined(__clang_major__) && __clang_major__ >= 23
    i32x8 z8 = {0, 0, 0, 0, 0, 0, 0, 0};
    __builtin_amdgcn_tensor_load_to_lds(g0, g1, z4, z4, z8, 0);
#else
    __builtin_amdgcn_tensor_load_to_lds(g0, g1, z4, z4, 0);
#endif
}
#endif

// =====================================================================
// f32 -> bf16 conversion
// =====================================================================
__global__ void f2bf_kernel(const float* __restrict__ in, bf16* __restrict__ out, size_t n) {
    size_t i = (size_t)blockIdx.x * blockDim.x + threadIdx.x;
    size_t stride = (size_t)gridDim.x * blockDim.x;
    for (; i < n; i += stride) out[i] = (bf16)in[i];
}

// =====================================================================
// Farthest point sampling: one block per (b, frame j); anchors use xyzs[b, 2j]
// =====================================================================
__global__ __launch_bounds__(256) void fps_kernel(const float* __restrict__ xyzs,
                                                  float* __restrict__ anchor) {
    int b = blockIdx.x / kTO;
    int j = blockIdx.x % kTO;
    const float* fr = xyzs + ((size_t)(b * kT + 2 * j)) * kN * 3;
    __shared__ float dist[kN];
    __shared__ float rv[256];
    __shared__ int   ri[256];
    __shared__ int   s_sel[kM];
    int tid = threadIdx.x;
    for (int n = tid; n < kN; n += 256) dist[n] = 1e10f;
    if (tid == 0) s_sel[0] = 0;
    __syncthreads();
    int last = 0;
    for (int it = 1; it < kM; it++) {
        float lx = fr[last * 3 + 0], ly = fr[last * 3 + 1], lz = fr[last * 3 + 2];
        float best = -1.0f; int bi = 0;
        for (int n = tid; n < kN; n += 256) {
            float dx = fr[n * 3 + 0] - lx;
            float dy = fr[n * 3 + 1] - ly;
            float dz = fr[n * 3 + 2] - lz;
            float d = dx * dx + dy * dy + dz * dz;
            float dn = fminf(dist[n], d);
            dist[n] = dn;
            if (dn > best) { best = dn; bi = n; }
        }
        rv[tid] = best; ri[tid] = bi;
        __syncthreads();
        for (int s2 = 128; s2 > 0; s2 >>= 1) {
            if (tid < s2) {
                if (rv[tid + s2] > rv[tid] ||
                    (rv[tid + s2] == rv[tid] && ri[tid + s2] < ri[tid])) {
                    rv[tid] = rv[tid + s2]; ri[tid] = ri[tid + s2];
                }
            }
            __syncthreads();
        }
        last = ri[0];
        if (tid == 0) s_sel[it] = last;
        __syncthreads();
    }
    for (int i = tid; i < kM; i += 256) {
        int n = s_sel[i];
        size_t o = (((size_t)(b * kTO) + j) * kM + i) * 3;
        anchor[o + 0] = fr[n * 3 + 0];
        anchor[o + 1] = fr[n * 3 + 1];
        anchor[o + 2] = fr[n * 3 + 2];
    }
}

// =====================================================================
// P4D conv: one block (512 threads) per anchor token.
// =====================================================================
__global__ __launch_bounds__(512) void p4dconv_kernel(const float* __restrict__ xyzs,
                                                      const float* __restrict__ oldf,
                                                      const float* __restrict__ wD,
                                                      const float* __restrict__ wF,
                                                      const float* __restrict__ anchor,
                                                      float* __restrict__ features) {
    int tok = blockIdx.x;                // (b*kTO + j)*kM + mi
    int mi = tok % kM;
    int bj = tok / kM;
    int j  = bj % kTO;
    int b  = bj / kTO;
    int tid = threadIdx.x;

    size_t ao = (((size_t)bj) * kM + mi) * 3;
    float ax = anchor[ao + 0], ay = anchor[ao + 1], az = anchor[ao + 2];

    __shared__ unsigned int flags[kN / 32];
    __shared__ int   s_idx[kNS];
    __shared__ float s_disp[kNS][4];
    __shared__ float s_feat[kNS][2];

    float wd0 = wD[tid * 4 + 0], wd1 = wD[tid * 4 + 1], wd2 = wD[tid * 4 + 2], wd3 = wD[tid * 4 + 3];
    float wf0 = wF[tid * 2 + 0], wf1 = wF[tid * 2 + 1];

    float pooled = -1e30f;
    for (int oo = 0; oo < 3; oo++) {
        int tp = 2 * j + oo;                 // padded-time index
        int src = (tp == 0) ? 0 : tp - 1;    // raw xyzs frame
        const float* nb   = xyzs + ((size_t)(b * kT + src)) * kN * 3;
        const float* fsrc = oldf + ((size_t)(b * kT + src)) * 2 * kN;

        for (int w = tid; w < kN / 32; w += 512) flags[w] = 0u;
        __syncthreads();
        for (int n = tid; n < kN; n += 512) {
            float dx = nb[n * 3 + 0] - ax;
            float dy = nb[n * 3 + 1] - ay;
            float dz = nb[n * 3 + 2] - az;
            if (dx * dx + dy * dy + dz * dz < kRAD2)
                atomicOr(&flags[n >> 5], 1u << (n & 31));
        }
        __syncthreads();
        if (tid == 0) {
            int cnt = 0;
            for (int w = 0; w < kN / 32 && cnt < kNS; w++) {
                unsigned int f = flags[w];
                while (f && cnt < kNS) {
                    int bit = __ffs(f) - 1;
                    s_idx[cnt++] = w * 32 + bit;
                    f &= f - 1;
                }
            }
            if (cnt == 0) { for (int k = 0; k < kNS; k++) s_idx[k] = 0; }
            else          { for (int k = cnt; k < kNS; k++) s_idx[k] = s_idx[0]; }
        }
        __syncthreads();
        if (tid < kNS) {
            int n = s_idx[tid];
            s_disp[tid][0] = nb[n * 3 + 0] - ax;
            s_disp[tid][1] = nb[n * 3 + 1] - ay;
            s_disp[tid][2] = nb[n * 3 + 2] - az;
            s_disp[tid][3] = (float)(oo - 1);
            s_feat[tid][0] = fsrc[n];
            s_feat[tid][1] = fsrc[kN + n];
        }
        __syncthreads();
        float best = -1e30f;
        #pragma unroll 4
        for (int k = 0; k < kNS; k++) {
            float de = wd0 * s_disp[k][0] + wd1 * s_disp[k][1] + wd2 * s_disp[k][2] + wd3 * s_disp[k][3];
            float fe = wf0 * s_feat[k][0] + wf1 * s_feat[k][1];
            best = fmaxf(best, de * fe);
        }
        pooled = fmaxf(pooled, best);
        __syncthreads();
    }
    features[(size_t)tok * kDIM + tid] = pooled;
}

// =====================================================================
// Token embedding
// =====================================================================
__global__ void embed_kernel(const float* __restrict__ anchor,
                             const float* __restrict__ features,
                             const float* __restrict__ pos_w, const float* __restrict__ pos_b,
                             const float* __restrict__ contr_w, const float* __restrict__ contr_b,
                             float* __restrict__ contr_out, float* __restrict__ x_out) {
    size_t total = (size_t)kB * kSEQ * kDIM;
    size_t i = (size_t)blockIdx.x * blockDim.x + threadIdx.x;
    size_t stride = (size_t)gridDim.x * blockDim.x;
    for (; i < total; i += stride) {
        int d   = (int)(i & (kDIM - 1));
        int tkn = (int)((i >> 9) % kSEQ);
        int b   = (int)((i >> 9) / kSEQ);
        int j   = tkn >> 7;
        int mi  = tkn & (kM - 1);
        size_t ao = (((size_t)(b * kTO) + j) * kM + mi) * 3;
        float c0 = anchor[ao], c1 = anchor[ao + 1], c2 = anchor[ao + 2];
        float c3 = (float)(j + 1);
        float pos = pos_w[d * 4] * c0 + pos_w[d * 4 + 1] * c1 + pos_w[d * 4 + 2] * c2 + pos_w[d * 4 + 3] * c3 + pos_b[d];
        float ctr = contr_w[d * 4] * c0 + contr_w[d * 4 + 1] * c1 + contr_w[d * 4 + 2] * c2 + contr_w[d * 4 + 3] * c3 + contr_b[d];
        contr_out[i] = ctr;
        x_out[i] = fmaxf(pos + features[i], 0.0f);
    }
}

// =====================================================================
// LayerNorm over DIM=512; bf16 (GEMM input) or f32 output
// =====================================================================
template <int OUTBF>
__global__ __launch_bounds__(256) void ln_kernel(const float* __restrict__ x,
                                                 const float* __restrict__ sc,
                                                 const float* __restrict__ bi,
                                                 void* __restrict__ out) {
    int row = blockIdx.x, tid = threadIdx.x;
    const float* xr = x + (size_t)row * kDIM;
    float v0 = xr[tid], v1 = xr[tid + 256];
    __shared__ float red[256];
    red[tid] = v0 + v1;
    __syncthreads();
    for (int s = 128; s > 0; s >>= 1) { if (tid < s) red[tid] += red[tid + s]; __syncthreads(); }
    float mu = red[0] * (1.0f / kDIM);
    __syncthreads();
    float d0 = v0 - mu, d1 = v1 - mu;
    red[tid] = d0 * d0 + d1 * d1;
    __syncthreads();
    for (int s = 128; s > 0; s >>= 1) { if (tid < s) red[tid] += red[tid + s]; __syncthreads(); }
    float rstd = rsqrtf(red[0] * (1.0f / kDIM) + 1e-5f);
    float o0 = d0 * rstd * sc[tid] + bi[tid];
    float o1 = d1 * rstd * sc[tid + 256] + bi[tid + 256];
    if (OUTBF) {
        bf16* o = (bf16*)out;
        o[(size_t)row * kDIM + tid] = (bf16)o0;
        o[(size_t)row * kDIM + tid + 256] = (bf16)o1;
    } else {
        float* o = (float*)out;
        o[(size_t)row * kDIM + tid] = o0;
        o[(size_t)row * kDIM + tid + 256] = o1;
    }
}

// =====================================================================
// WMMA GEMM: C[M,N] = epi(A[M,K](bf16) @ B[N,K]^T(bf16) + bias) (+res)
// 256 threads = 8 waves; each wave computes a 32x32 output tile (4 WMMA accs,
// A/B fragments each reused twice). Block tile = 128x64.
// ACT: 0=none, 1=relu, 2=gelu.  OUTBF: 1 -> bf16 out, 0 -> f32 out
// =====================================================================
template <int ACT, int OUTBF>
__global__ __launch_bounds__(256) void wmma_gemm_kernel(const bf16* __restrict__ A,
                                                        const bf16* __restrict__ Bw,
                                                        const float* __restrict__ bias,
                                                        const float* __restrict__ res,
                                                        void* __restrict__ out,
                                                        int Nn, int K) {
    int wave = threadIdx.x >> 5;
    int lane = threadIdx.x & 31;
    int lrow = lane & 15;
    int hsel = lane >> 4;
    int rbase = blockIdx.x * 128 + (wave >> 1) * 32;
    int cbase = blockIdx.y * 64 + (wave & 1) * 32;

    v8f c00 = {}, c01 = {}, c10 = {}, c11 = {};
    const bf16* arow0 = A + (size_t)(rbase + lrow) * K;
    const bf16* arow1 = A + (size_t)(rbase + 16 + lrow) * K;
    const bf16* brow0 = Bw + (size_t)(cbase + lrow) * K;
    const bf16* brow1 = Bw + (size_t)(cbase + 16 + lrow) * K;

    for (int kk = 0; kk < K; kk += 32) {
        __builtin_prefetch(arow0 + kk + 128, 0, 0);
        __builtin_prefetch(arow1 + kk + 128, 0, 0);
        v16bf a0, a1;
        {
            const bf16* p0 = arow0 + kk + (hsel << 4);
            const bf16* p1 = arow1 + kk + (hsel << 4);
            v8bf x0 = *(const v8bf*)p0, x1 = *(const v8bf*)(p0 + 8);
            v8bf y0 = *(const v8bf*)p1, y1 = *(const v8bf*)(p1 + 8);
            #pragma unroll
            for (int i = 0; i < 8; i++) {
                a0[i] = x0[i]; a0[8 + i] = x1[i];
                a1[i] = y0[i]; a1[8 + i] = y1[i];
            }
        }
        v16bf b0 = *(const v16bf*)(brow0 + kk + (hsel << 4));
        v16bf b1 = *(const v16bf*)(brow1 + kk + (hsel << 4));
        c00 = __builtin_amdgcn_wmma_f32_16x16x32_bf16(false, a0, false, b0, (short)0, c00, false, false);
        c01 = __builtin_amdgcn_wmma_f32_16x16x32_bf16(false, a0, false, b1, (short)0, c01, false, false);
        c10 = __builtin_amdgcn_wmma_f32_16x16x32_bf16(false, a1, false, b0, (short)0, c10, false, false);
        c11 = __builtin_amdgcn_wmma_f32_16x16x32_bf16(false, a1, false, b1, (short)0, c11, false, false);
    }

    #pragma unroll
    for (int rt = 0; rt < 2; rt++) {
        #pragma unroll
        for (int ct = 0; ct < 2; ct++) {
            v8f c = (rt == 0) ? (ct == 0 ? c00 : c01) : (ct == 0 ? c10 : c11);
            #pragma unroll
            for (int j = 0; j < 8; j++) {
                int row = rbase + rt * 16 + j + hsel * 8;
                int col = cbase + ct * 16 + lrow;
                float v = c[j];
                if (bias) v += bias[col];
                if (ACT == 1) v = fmaxf(v, 0.0f);
                if (ACT == 2) v = gelu_exact(v);
                if (res) v += res[(size_t)row * Nn + col];
                if (OUTBF) ((bf16*)out)[(size_t)row * Nn + col] = (bf16)v;
                else       ((float*)out)[(size_t)row * Nn + col] = v;
            }
        }
    }
}

// =====================================================================
// Flash attention: qkv bf16 [B*SEQ, 3*512] (q|k|v), out bf16 [B*SEQ, 512].
// Block = 8 waves over 128 query rows of one (b,h); K/V tiles (32x64 bf16)
// are staged into LDS once per block via the Tensor Data Mover, softmax row
// reductions run on v_permlane16.
// =====================================================================
__global__ __launch_bounds__(256) void attn_kernel(const bf16* __restrict__ qkv,
                                                   bf16* __restrict__ o_out) {
    int bid = blockIdx.x;
    int qt = bid % (kSEQ / 128); bid /= (kSEQ / 128);
    int h  = bid % kHEADS;
    int b  = bid / kHEADS;
    int wave = threadIdx.x >> 5;
    int lane = threadIdx.x & 31;
    int lrow = lane & 15;
    int hsel = lane >> 4;
    int qbase = qt * 128 + wave * 16;
    const bf16* base = qkv + (size_t)b * kSEQ * 1536;
    const float scale = 0.125f;           // 64^-0.5

    __shared__ __align__(64) bf16 sK[32][64];
    __shared__ __align__(64) bf16 sV[32][64];
    __shared__ bf16 lds_p[8][16][32];

    // Q operand fragments for the two K=32 steps over head dim 64
    v16bf aq[2];
    {
        const bf16* qp = base + (size_t)(qbase + lrow) * 1536 + h * kDH;
        #pragma unroll
        for (int s = 0; s < 2; s++) {
            const bf16* p = qp + s * 32 + hsel * 16;
            v8bf x0 = *(const v8bf*)p;
            v8bf x1 = *(const v8bf*)(p + 8);
            #pragma unroll
            for (int i = 0; i < 8; i++) { aq[s][i] = x0[i]; aq[s][8 + i] = x1[i]; }
        }
    }

    float m_r[8], l_r[8];
    #pragma unroll
    for (int j = 0; j < 8; j++) { m_r[j] = -1e30f; l_r[j] = 0.0f; }
    v8f oacc[4] = {};

    for (int kt = 0; kt < kSEQ / 32; kt++) {
        int kb = kt * 32;
        // ---- stage K/V 32x64 tiles into LDS ----
        __syncthreads();   // previous tile fully consumed
#if HAVE_TDM
        if (threadIdx.x == 0) {
            tdm_load_2d(base + (size_t)kb * 1536 + 512 + h * kDH,
                        (unsigned)(size_t)&sK[0][0], 32u, 64u, 1536u);
            tdm_load_2d(base + (size_t)kb * 1536 + 1024 + h * kDH,
                        (unsigned)(size_t)&sV[0][0], 32u, 64u, 1536u);
        }
        if (threadIdx.x < 32) __builtin_amdgcn_s_wait_tensorcnt(0);
#else
        for (int i = threadIdx.x; i < 32 * 64; i += 256) {
            int r = i >> 6, c = i & 63;
            sK[r][c] = base[(size_t)(kb + r) * 1536 + 512 + h * kDH + c];
            sV[r][c] = base[(size_t)(kb + r) * 1536 + 1024 + h * kDH + c];
        }
#endif
        __syncthreads();

        // ---- S = Q @ K^T for this 32-key tile ----
        v8f s0 = {}, s1 = {};
        #pragma unroll
        for (int s = 0; s < 2; s++) {
            v16bf bk0 = *(const v16bf*)&sK[lrow][s * 32 + hsel * 16];
            v16bf bk1 = *(const v16bf*)&sK[16 + lrow][s * 32 + hsel * 16];
            s0 = __builtin_amdgcn_wmma_f32_16x16x32_bf16(false, aq[s], false, bk0, (short)0, s0, false, false);
            s1 = __builtin_amdgcn_wmma_f32_16x16x32_bf16(false, aq[s], false, bk1, (short)0, s1, false, false);
        }
        // ---- online softmax per owned row (row = j + 8*hsel) ----
        float corr[8];
        #pragma unroll
        for (int j = 0; j < 8; j++) {
            float v0 = s0[j] * scale, v1 = s1[j] * scale;
            float rm = redmax16(fmaxf(v0, v1));
            float mn = fmaxf(m_r[j], rm);
            corr[j] = __expf(m_r[j] - mn);
            float p0 = __expf(v0 - mn);
            float p1 = __expf(v1 - mn);
            float rs = redsum16(p0 + p1);
            l_r[j] = l_r[j] * corr[j] + rs;
            m_r[j] = mn;
            int r = j + hsel * 8;
            lds_p[wave][r][lrow]      = (bf16)p0;
            lds_p[wave][r][16 + lrow] = (bf16)p1;
        }
        #pragma unroll
        for (int t = 0; t < 4; t++)
            #pragma unroll
            for (int j = 0; j < 8; j++) oacc[t][j] *= corr[j];
        __builtin_amdgcn_wave_barrier();
        // ---- reload P as an A-operand fragment (16x32) ----
        v16bf pa;
        {
            int ks = hsel * 16;
            #pragma unroll
            for (int i = 0; i < 8; i++) {
                pa[i]     = lds_p[wave][lrow][ks + i];
                pa[8 + i] = lds_p[wave][lrow][ks + 8 + i];
            }
        }
        // ---- O += P @ V (V operand gathered from LDS tile) ----
        #pragma unroll
        for (int t = 0; t < 4; t++) {
            int dcol = t * 16 + lrow;
            v16bf bv;
            #pragma unroll
            for (int kk2 = 0; kk2 < 16; kk2++) bv[kk2] = sV[hsel * 16 + kk2][dcol];
            oacc[t] = __builtin_amdgcn_wmma_f32_16x16x32_bf16(false, pa, false, bv, (short)0, oacc[t], false, false);
        }
    }

    #pragma unroll
    for (int t = 0; t < 4; t++)
        #pragma unroll
        for (int j = 0; j < 8; j++) {
            int row = qbase + j + hsel * 8;
            int d = t * 16 + lrow;
            float v = oacc[t][j] / l_r[j];
            o_out[((size_t)b * kSEQ + row) * kDIM + h * kDH + d] = (bf16)v;
        }
}

// =====================================================================
// Max-pool over tokens, head LN (reuses ln_kernel), head MLP
// =====================================================================
__global__ __launch_bounds__(512) void pool_kernel(const float* __restrict__ x, float* __restrict__ pooled) {
    int b = blockIdx.x, d = threadIdx.x;
    float m = -1e30f;
    for (int t = 0; t < kSEQ; t++)
        m = fmaxf(m, x[((size_t)b * kSEQ + t) * kDIM + d]);
    pooled[b * kDIM + d] = m;
}

__global__ void head1_kernel(const float* __restrict__ hln, const float* __restrict__ w,
                             const float* __restrict__ b1, float* __restrict__ h1) {
    int o = blockIdx.x * blockDim.x + threadIdx.x;   // 4 * 1024
    if (o >= kB * kMLP) return;
    int b = o >> 10, oc = o & (kMLP - 1);
    const float* hr = hln + (size_t)b * kDIM;
    const float* wr = w + (size_t)oc * kDIM;
    float acc = b1[oc];
    for (int k = 0; k < kDIM; k++) acc += hr[k] * wr[k];
    h1[o] = gelu_exact(acc);
}

__global__ void head2_kernel(const float* __restrict__ h1, const float* __restrict__ w,
                             const float* __restrict__ b2, float* __restrict__ logits) {
    int t = threadIdx.x;
    if (t >= kB * kNC) return;
    int b = t / kNC, o = t % kNC;
    const float* hr = h1 + (size_t)b * kMLP;
    const float* wr = w + (size_t)o * kMLP;
    float acc = b2[o];
    for (int k = 0; k < kMLP; k++) acc += hr[k] * wr[k];
    logits[t] = acc;
}

// =====================================================================
// host launch
// =====================================================================
extern "C" void kernel_launch(void* const* d_in, const int* in_sizes, int n_in,
                              void* d_out, int out_size, void* d_ws, size_t ws_size,
                              hipStream_t stream) {
    const float* xyzs    = (const float*)d_in[0];
    const float* oldf    = (const float*)d_in[1];
    const float* conv_dw = (const float*)d_in[2];
    const float* conv_fw = (const float*)d_in[3];
    const float* pos_w   = (const float*)d_in[4];
    const float* pos_b   = (const float*)d_in[5];
    const float* contr_w = (const float*)d_in[6];
    const float* contr_b = (const float*)d_in[7];
    const float* ln1_s   = (const float*)d_in[8];
    const float* ln1_b   = (const float*)d_in[9];
    const float* qkv_w   = (const float*)d_in[10];
    const float* out_w   = (const float*)d_in[11];
    const float* out_b   = (const float*)d_in[12];
    const float* ln2_s   = (const float*)d_in[13];
    const float* ln2_b   = (const float*)d_in[14];
    const float* ff1_w   = (const float*)d_in[15];
    const float* ff1_b   = (const float*)d_in[16];
    const float* ff2_w   = (const float*)d_in[17];
    const float* ff2_b   = (const float*)d_in[18];
    const float* hln_s   = (const float*)d_in[19];
    const float* hln_b   = (const float*)d_in[20];
    const float* h1_w    = (const float*)d_in[21];
    const float* h1_b    = (const float*)d_in[22];
    const float* h2_w    = (const float*)d_in[23];
    const float* h2_b    = (const float*)d_in[24];

    float* out_logits = (float*)d_out;
    float* out_contr  = out_logits + kB * kNC;
    float* out_feat   = out_contr + (size_t)kB * kSEQ * kDIM;

    const int MTOK = kB * kSEQ;          // 6144 token rows
    char* ws = (char*)d_ws;
    size_t cur = 0;
    auto alloc = [&](size_t bytes) -> char* {
        char* p = ws + cur;
        cur = (cur + bytes + 255) & ~(size_t)255;
        return p;
    };
    bf16*  qkvw_bf = (bf16*)alloc((size_t)kDEPTH * 3 * kDIM * kDIM * sizeof(bf16));
    bf16*  outw_bf = (bf16*)alloc((size_t)kDEPTH * kDIM * kDIM * sizeof(bf16));
    bf16*  ff1w_bf = (bf16*)alloc((size_t)kDEPTH * kMLP * kDIM * sizeof(bf16));
    bf16*  ff2w_bf = (bf16*)alloc((size_t)kDEPTH * kDIM * kMLP * sizeof(bf16));
    float* anchor  = (float*)alloc((size_t)kB * kTO * kM * 3 * sizeof(float));
    float* x_buf   = (float*)alloc((size_t)MTOK * kDIM * sizeof(float));
    bf16*  h_bf    = (bf16*)alloc((size_t)MTOK * kDIM * sizeof(bf16));
    bf16*  qkv_act = (bf16*)alloc((size_t)MTOK * 3 * kDIM * sizeof(bf16)); // reused for ff1 act
    bf16*  o_bf    = (bf16*)alloc((size_t)MTOK * kDIM * sizeof(bf16));
    float* pooled  = (float*)alloc((size_t)kB * kDIM * sizeof(float));
    float* hln     = (float*)alloc((size_t)kB * kDIM * sizeof(float));
    float* h1_act  = (float*)alloc((size_t)kB * kMLP * sizeof(float));
    (void)ws_size; (void)n_in; (void)in_sizes; (void)out_size;

    auto cdiv = [](size_t a, size_t b) { return (a + b - 1) / b; };

    // weight conversions (deterministic, every call)
    {
        size_t n;
        n = (size_t)kDEPTH * 3 * kDIM * kDIM;
        f2bf_kernel<<<(unsigned)cdiv(n, 256), 256, 0, stream>>>(qkv_w, qkvw_bf, n);
        n = (size_t)kDEPTH * kDIM * kDIM;
        f2bf_kernel<<<(unsigned)cdiv(n, 256), 256, 0, stream>>>(out_w, outw_bf, n);
        n = (size_t)kDEPTH * kMLP * kDIM;
        f2bf_kernel<<<(unsigned)cdiv(n, 256), 256, 0, stream>>>(ff1_w, ff1w_bf, n);
        n = (size_t)kDEPTH * kDIM * kMLP;
        f2bf_kernel<<<(unsigned)cdiv(n, 256), 256, 0, stream>>>(ff2_w, ff2w_bf, n);
    }

    // stage 1: P4D conv
    fps_kernel<<<kB * kTO, 256, 0, stream>>>(xyzs, anchor);
    p4dconv_kernel<<<kB * kTO * kM, 512, 0, stream>>>(xyzs, oldf, conv_dw, conv_fw, anchor, out_feat);
    {
        size_t n = (size_t)MTOK * kDIM;
        embed_kernel<<<(unsigned)cdiv(n, 256), 256, 0, stream>>>(anchor, out_feat, pos_w, pos_b,
                                                                 contr_w, contr_b, out_contr, x_buf);
    }

    // stage 2: transformer
    for (int l = 0; l < kDEPTH; l++) {
        ln_kernel<1><<<MTOK, 256, 0, stream>>>(x_buf, ln1_s + l * kDIM, ln1_b + l * kDIM, (void*)h_bf);
        wmma_gemm_kernel<0, 1><<<dim3(MTOK / 128, (3 * kDIM) / 64), 256, 0, stream>>>(
            h_bf, qkvw_bf + (size_t)l * 3 * kDIM * kDIM, nullptr, nullptr, (void*)qkv_act,
            3 * kDIM, kDIM);
        attn_kernel<<<kB * kHEADS * (kSEQ / 128), 256, 0, stream>>>(qkv_act, o_bf);
        wmma_gemm_kernel<0, 0><<<dim3(MTOK / 128, kDIM / 64), 256, 0, stream>>>(
            o_bf, outw_bf + (size_t)l * kDIM * kDIM, out_b + l * kDIM, x_buf, (void*)x_buf,
            kDIM, kDIM);
        ln_kernel<1><<<MTOK, 256, 0, stream>>>(x_buf, ln2_s + l * kDIM, ln2_b + l * kDIM, (void*)h_bf);
        wmma_gemm_kernel<2, 1><<<dim3(MTOK / 128, kMLP / 64), 256, 0, stream>>>(
            h_bf, ff1w_bf + (size_t)l * kMLP * kDIM, ff1_b + l * kMLP, nullptr, (void*)qkv_act,
            kMLP, kDIM);
        wmma_gemm_kernel<0, 0><<<dim3(MTOK / 128, kDIM / 64), 256, 0, stream>>>(
            qkv_act, ff2w_bf + (size_t)l * kDIM * kMLP, ff2_b + l * kDIM, x_buf, (void*)x_buf,
            kDIM, kMLP);
    }

    // stage 3: head
    pool_kernel<<<kB, kDIM, 0, stream>>>(x_buf, pooled);
    ln_kernel<0><<<kB, 256, 0, stream>>>(pooled, hln_s, hln_b, (void*)hln);
    head1_kernel<<<(kB * kMLP + 255) / 256, 256, 0, stream>>>(hln, h1_w, h1_b, h1_act);
    head2_kernel<<<1, 128, 0, stream>>>(h1_act, h2_w, h2_b, out_logits);
}
